// Model_39676907884648
// MI455X (gfx1250) — matrix operand order repaired and checked by reading, not compiled, not run on hardware
//
#include <hip/hip_runtime.h>
#include <math.h>

typedef __attribute__((ext_vector_type(16))) __bf16 v16bf;
typedef __attribute__((ext_vector_type(8)))  float  v8f;

__device__ __forceinline__ unsigned short f32_to_bf16(float f) {
    unsigned int u = __float_as_uint(f);
    u += 0x7FFFu + ((u >> 16) & 1u);   // round-to-nearest-even
    return (unsigned short)(u >> 16);
}

// ---------------- elementwise f32 -> bf16 convert (vectorized x4) ----------------
__global__ __launch_bounds__(256) void convert_f32_bf16(const float* __restrict__ in,
                                                        unsigned short* __restrict__ out,
                                                        size_t n) {
    size_t i = ((size_t)blockIdx.x * 256 + threadIdx.x) * 4;
    if (i + 3 < n) {
        float4 f = *(const float4*)(in + i);
        ushort4 o;
        o.x = f32_to_bf16(f.x); o.y = f32_to_bf16(f.y);
        o.z = f32_to_bf16(f.z); o.w = f32_to_bf16(f.w);
        *(ushort4*)(out + i) = o;
    }
}

// ---------------- batched bf16 transpose: out[b][c][r] = in[b][r][c] ----------------
__global__ __launch_bounds__(256) void transpose_bf16(const unsigned short* __restrict__ in,
                                                      unsigned short* __restrict__ out,
                                                      int rows, int cols) {
    __shared__ unsigned short t[32][33];
    const size_t boff = (size_t)blockIdx.z * rows * cols;
    const int c0 = blockIdx.x * 32, r0 = blockIdx.y * 32;
    const int tx = threadIdx.x & 31, ty = threadIdx.x >> 5;
    #pragma unroll
    for (int i = 0; i < 32; i += 8)
        t[ty + i][tx] = in[boff + (size_t)(r0 + ty + i) * cols + (c0 + tx)];
    __syncthreads();
    #pragma unroll
    for (int i = 0; i < 32; i += 8)
        out[boff + (size_t)(c0 + ty + i) * rows + (r0 + tx)] = t[tx][ty + i];
}

// ============ TN bf16 WMMA GEMM: C[M,N] = scale * (A[MxK] . B'[NxK]^T) ============
// Both operands row-major with K contiguous -> identical coalesced staging for A and B'.
// Block tile 128(M) x 128(N), K-step 32, double-buffered LDS, one barrier per K-step.
// 8 waves (wave32): wave (wm,wn) owns a 32x64 patch = 2x4 WMMA tiles (64 f32 acc VGPRs).
template <bool OUT_BF16>
__global__ __launch_bounds__(256) void gemm_tn_bf16_wmma(const unsigned short* __restrict__ A,
                                                         const unsigned short* __restrict__ Bp,
                                                         void* __restrict__ Cm,
                                                         int M, int N, int K,
                                                         long long aBatch, long long bBatch, long long cBatch,
                                                         float scale) {
    constexpr int LDT = 40;                     // padded LDS pitch: 32 data + 8 pad halfwords (80B rows)
    __shared__ unsigned short lA[2][128 * LDT]; // A tile: [row][k]
    __shared__ unsigned short lB[2][128 * LDT]; // B' tile: [col][k]

    const int tid  = threadIdx.x;
    const int lane = tid & 31;
    const int wave = tid >> 5;
    const int wm   = wave >> 1;                 // 0..3 -> 32-row group
    const int wn   = wave & 1;                  // 0..1 -> 64-col group

    A  += (size_t)blockIdx.z * aBatch;
    Bp += (size_t)blockIdx.z * bBatch;
    const size_t cOff = (size_t)blockIdx.z * cBatch;

    const int block_row = blockIdx.y * 128;
    const int block_col = blockIdx.x * 128;

    // staging: thread -> tile row (tid>>1), K half (tid&1)*16; 2x b128 per operand
    const int s_r = tid >> 1;
    const int s_k = (tid & 1) * 16;
    const unsigned short* aPtr = A  + (size_t)(block_row + s_r) * K + s_k;
    const unsigned short* bPtr = Bp + (size_t)(block_col + s_r) * K + s_k;

    v8f acc[2][4];
    #pragma unroll
    for (int m = 0; m < 2; ++m)
        #pragma unroll
        for (int n = 0; n < 4; ++n)
            #pragma unroll
            for (int e = 0; e < 8; ++e)
                acc[m][n][e] = 0.0f;

    union Frag { struct { uint4 lo, hi; } q; v16bf v; };

    const int rowA = lane & 15;
    const int kA0  = (lane < 16) ? 0 : 8;    // 16-bit A layout: lanes 0-15 K0-7/16-23, lanes 16-31 K8-15/24-31
    const int kB0  = (lane < 16) ? 0 : 8;    // 16-bit B layout: lanes 0-15 K0-7/16-23, lanes 16-31 K8-15/24-31

    // ---- prologue: stage K-tile 0 into buffer 0 ----
    {
        uint4 a0 = *(const uint4*)(aPtr);
        uint4 a1 = *(const uint4*)(aPtr + 8);
        uint4 b0 = *(const uint4*)(bPtr);
        uint4 b1 = *(const uint4*)(bPtr + 8);
        uint4* da = (uint4*)&lA[0][s_r * LDT + s_k];
        uint4* db = (uint4*)&lB[0][s_r * LDT + s_k];
        da[0] = a0; da[1] = a1;
        db[0] = b0; db[1] = b1;
    }
    __syncthreads();

    const int kTiles = K >> 5;
    for (int kt = 0; kt < kTiles; ++kt) {
        const int  cur  = kt & 1;
        const bool more = (kt + 1) < kTiles;

        // ---- issue global loads for the next K-tile while WMMAs consume this one ----
        uint4 na0, na1, nb0, nb1;
        if (more) {
            const unsigned short* ap = aPtr + (size_t)(kt + 1) * 32;
            const unsigned short* bp = bPtr + (size_t)(kt + 1) * 32;
            na0 = *(const uint4*)(ap); na1 = *(const uint4*)(ap + 8);
            nb0 = *(const uint4*)(bp); nb1 = *(const uint4*)(bp + 8);
            if (kt + 2 < kTiles) {
                __builtin_prefetch(aPtr + (size_t)(kt + 2) * 32, 0, 3);
                __builtin_prefetch(bPtr + (size_t)(kt + 2) * 32, 0, 3);
            }
        }

        // ---- fragments from LDS (all 16B-aligned ds_load_b128) ----
        Frag af[2], bfm[4];
        #pragma unroll
        for (int m = 0; m < 2; ++m) {
            const unsigned short* p = &lA[cur][(wm * 32 + m * 16 + rowA) * LDT + kA0];
            af[m].q.lo = *(const uint4*)(p);
            af[m].q.hi = *(const uint4*)(p + 16);
        }
        #pragma unroll
        for (int n = 0; n < 4; ++n) {
            const unsigned short* p = &lB[cur][(wn * 64 + n * 16 + rowA) * LDT + kB0];
            bfm[n].q.lo = *(const uint4*)(p);
            bfm[n].q.hi = *(const uint4*)(p + 16);
        }

        // ---- 8 WMMAs advance the 32x64 patch by K=32 ----
        #pragma unroll
        for (int m = 0; m < 2; ++m)
            #pragma unroll
            for (int n = 0; n < 4; ++n)
                acc[m][n] = __builtin_amdgcn_wmma_f32_16x16x32_bf16(
                    false, af[m].v, false, bfm[n].v, (short)0, acc[m][n], false, false);

        // ---- stash next tile into the alternate buffer ----
        if (more) {
            uint4* da = (uint4*)&lA[cur ^ 1][s_r * LDT + s_k];
            uint4* db = (uint4*)&lB[cur ^ 1][s_r * LDT + s_k];
            da[0] = na0; da[1] = na1;
            db[0] = nb0; db[1] = nb1;
        }
        __syncthreads();
    }

    // ---- writeback: C 16x16 f32 layout (VGPR i -> row i / i+8; lanes 0-15 / 16-31 cols) ----
    const int colBase = block_col + wn * 64 + (lane & 15);
    #pragma unroll
    for (int m = 0; m < 2; ++m) {
        const int rbase = block_row + wm * 32 + m * 16 + ((lane < 16) ? 0 : 8);
        #pragma unroll
        for (int n = 0; n < 4; ++n) {
            const int col = colBase + n * 16;
            #pragma unroll
            for (int i = 0; i < 8; ++i) {
                float v = acc[m][n][i] * scale;
                size_t idx = cOff + (size_t)(rbase + i) * N + col;
                if constexpr (OUT_BF16) ((unsigned short*)Cm)[idx] = f32_to_bf16(v);
                else                    ((float*)Cm)[idx] = v;
            }
        }
    }
}

// ---------------- row softmax: fp32 in, bf16 out (float4 paths) ----------------
__global__ __launch_bounds__(256) void softmax_rows(const float* __restrict__ in,
                                                    unsigned short* __restrict__ out,
                                                    int cols) {
    __shared__ float red[256];
    const size_t row = blockIdx.x;
    const float4* r4 = (const float4*)(in + row * cols);
    const int n4 = cols >> 2;

    float lmax = -INFINITY;
    for (int c = threadIdx.x; c < n4; c += 256) {
        float4 f = r4[c];
        lmax = fmaxf(lmax, fmaxf(fmaxf(f.x, f.y), fmaxf(f.z, f.w)));
    }
    red[threadIdx.x] = lmax; __syncthreads();
    for (int s = 128; s > 0; s >>= 1) {
        if (threadIdx.x < s) red[threadIdx.x] = fmaxf(red[threadIdx.x], red[threadIdx.x + s]);
        __syncthreads();
    }
    const float m = red[0]; __syncthreads();

    float lsum = 0.f;
    for (int c = threadIdx.x; c < n4; c += 256) {
        float4 f = r4[c];
        lsum += __expf(f.x - m) + __expf(f.y - m) + __expf(f.z - m) + __expf(f.w - m);
    }
    red[threadIdx.x] = lsum; __syncthreads();
    for (int s = 128; s > 0; s >>= 1) {
        if (threadIdx.x < s) red[threadIdx.x] += red[threadIdx.x + s];
        __syncthreads();
    }
    const float inv = 1.f / red[0];

    ushort4* w4 = (ushort4*)(out + row * cols);
    for (int c = threadIdx.x; c < n4; c += 256) {
        float4 f = r4[c];
        ushort4 o;
        o.x = f32_to_bf16(__expf(f.x - m) * inv);
        o.y = f32_to_bf16(__expf(f.y - m) * inv);
        o.z = f32_to_bf16(__expf(f.z - m) * inv);
        o.w = f32_to_bf16(__expf(f.w - m) * inv);
        w4[c] = o;
    }
}

// =====================================================================================
extern "C" void kernel_launch(void* const* d_in, const int* in_sizes, int n_in,
                              void* d_out, int out_size, void* d_ws, size_t ws_size,
                              hipStream_t stream) {
    (void)in_sizes; (void)n_in; (void)out_size; (void)ws_size;
    constexpr int Bn = 4, S = 2048, D = 2048;
    constexpr size_t SD = (size_t)S * D;
    constexpr size_t SS = (size_t)S * S;

    const float* xq = (const float*)d_in[0];
    const float* xk = (const float*)d_in[1];
    // d_in[2] (xv), d_in[3] (attn_mask), d_in[6] (Wv): dead code in the reference
    const float* Wq = (const float*)d_in[4];
    const float* Wk = (const float*)d_in[5];

    size_t off = 0;
    auto alloc = [&](size_t bytes) {
        void* p = (char*)d_ws + off;
        off += (bytes + 255) & ~(size_t)255;
        return p;
    };
    unsigned short* xq_bf  = (unsigned short*)alloc(Bn * SD * 2);
    unsigned short* xk_bf  = (unsigned short*)alloc(Bn * SD * 2);
    unsigned short* Wq_bf  = (unsigned short*)alloc((size_t)D * D * 2);
    unsigned short* Wk_bf  = (unsigned short*)alloc((size_t)D * D * 2);
    unsigned short* q_bf   = (unsigned short*)alloc(Bn * SD * 2);
    unsigned short* k_bf   = (unsigned short*)alloc(Bn * SD * 2);
    unsigned short* att_bf = (unsigned short*)alloc(Bn * SS * 2);
    unsigned short* attT   = (unsigned short*)alloc(Bn * SS * 2);
    float*          qk_f   = (float*)alloc(Bn * SS * 4);

    // 1) convert everything live to bf16 (weights used as-is: W[e,d] is already NxK for TN GEMM)
    convert_f32_bf16<<<(unsigned)(Bn * SD / 1024), 256, 0, stream>>>(xq, xq_bf, Bn * SD);
    convert_f32_bf16<<<(unsigned)(Bn * SD / 1024), 256, 0, stream>>>(xk, xk_bf, Bn * SD);
    convert_f32_bf16<<<(unsigned)((size_t)D * D / 1024), 256, 0, stream>>>(Wq, Wq_bf, (size_t)D * D);
    convert_f32_bf16<<<(unsigned)((size_t)D * D / 1024), 256, 0, stream>>>(Wk, Wk_bf, (size_t)D * D);

    const dim3 gGemm(S / 128, S / 128, Bn);
    // 2) q = xq . Wq^T   (TN: B' = Wq [e,d], bf16 out)
    gemm_tn_bf16_wmma<true><<<gGemm, 256, 0, stream>>>(xq_bf, Wq_bf, q_bf, S, D, D,
                                                       (long long)SD, 0, (long long)SD, 1.0f);
    // 3) k = xk . Wk^T
    gemm_tn_bf16_wmma<true><<<gGemm, 256, 0, stream>>>(xk_bf, Wk_bf, k_bf, S, D, D,
                                                       (long long)SD, 0, (long long)SD, 1.0f);
    // 4) qk = (q . k^T) / sqrt(D)  (TN: B' = k [t,d], fp32 out for softmax)
    gemm_tn_bf16_wmma<false><<<gGemm, 256, 0, stream>>>(q_bf, k_bf, qk_f, S, S, D,
                                                        (long long)SD, (long long)SD, (long long)SS,
                                                        1.0f / sqrtf((float)D));
    // 5) attn = softmax(qk) rows (bf16 out)
    softmax_rows<<<Bn * S, 256, 0, stream>>>(qk_f, att_bf, S);
    // 6) attn^T so the last GEMM is also TN
    {
        dim3 g(S / 32, S / 32, Bn);
        transpose_bf16<<<g, 256, 0, stream>>>(att_bf, attT, S, S);
    }
    // 7) out[i,j] = sum_m k[i,m] attn[m,j]  (TN: A = k, B' = attn^T [j,m], fp32 out)
    gemm_tn_bf16_wmma<false><<<gGemm, 256, 0, stream>>>(k_bf, attT, (float*)d_out, S, S, D,
                                                        (long long)SD, (long long)SS, (long long)SS,
                                                        1.0f);
}
